// LightweightMamba2MoE_66116726555023
// MI455X (gfx1250) — compile-verified, matrix operand order checked by reading, not run on hardware
//
#include <hip/hip_runtime.h>
#include <hip/hip_bf16.h>
#include <math.h>

// ---------------------------------------------------------------------------
// Problem constants: B=8 L=2048 D=1024 E=8 H=2048
// ---------------------------------------------------------------------------
constexpr int kD = 1024;
constexpr int kE = 8;
constexpr int kH = 2048;
constexpr int kN = 8 * 2048;     // 16384 tokens
constexpr int MT = 32;           // tokens per expert workgroup (2 x 16-row tiles)
constexpr int HPAD = kH + 8;     // hidden row stride (bf16), keeps 16B alignment
constexpr int DPAD = kD + 8;     // x-tile row stride (bf16), keeps 16B alignment

typedef __attribute__((ext_vector_type(16))) __bf16 v16bf;
typedef __attribute__((ext_vector_type(8)))  float  v8f;
typedef __attribute__((__vector_size__(16))) int    i32x4;

// Workspace layout
constexpr size_t WS_META  = 256 + (size_t)kN * 4 + (size_t)kE * kN * 4;  // 590,080 B
constexpr size_t WS_XB    = (size_t)kN * kD * 2;                          // 32 MB
constexpr size_t WS_W     = (size_t)kE * kD * kH * 2;                     // 32 MB each
constexpr size_t WS_TOTAL = WS_META + WS_XB + 2 * WS_W;                   // ~96.6 MB

#if __has_builtin(__builtin_amdgcn_global_load_async_to_lds_b128)
#define ASYNC_LDS 1
#define AS1 __attribute__((address_space(1)))
#define AS3 __attribute__((address_space(3)))
#endif

// ---------------------------------------------------------------------------
// Fragment loaders per CDNA5 WMMA 16-bit striping:
//   lane l: row/col = l & 15, hi = l >> 4
//   elems j=0..7  -> K = kbase + hi*8 + j
//   elems j=8..15 -> K = kbase + 16 + hi*8 + (j-8)
// With bf16 data contiguous along K this is two 16-byte loads.
// ---------------------------------------------------------------------------
__device__ __forceinline__ v16bf load_frag_bf16(const __bf16* __restrict__ p) {
  v16bf a;
#pragma unroll
  for (int j = 0; j < 8; ++j) a[j] = p[j];
#pragma unroll
  for (int j = 0; j < 8; ++j) a[8 + j] = p[16 + j];
  return a;
}

__device__ __forceinline__ v16bf load_a_frag_f32(const float* __restrict__ p) {
  v16bf a;
#pragma unroll
  for (int j = 0; j < 8; ++j) a[j] = (__bf16)p[j];
#pragma unroll
  for (int j = 0; j < 8; ++j) a[8 + j] = (__bf16)p[16 + j];
  return a;
}

__device__ __forceinline__ v16bf load_b_frag_f32(const float* __restrict__ wcol, int ld) {
  v16bf b;
#pragma unroll
  for (int j = 0; j < 8; ++j) b[j] = (__bf16)wcol[(size_t)j * ld];
  const float* wcol2 = wcol + (size_t)16 * ld;
#pragma unroll
  for (int j = 0; j < 8; ++j) b[8 + j] = (__bf16)wcol2[(size_t)j * ld];
  return b;
}

// ---------------------------------------------------------------------------
// Kernel 0: zero per-expert counters
// ---------------------------------------------------------------------------
__global__ void moe_init_kernel(int* __restrict__ counts) {
  if (threadIdx.x < kE) counts[threadIdx.x] = 0;
}

// ---------------------------------------------------------------------------
// Kernel 1: gating (one wave32 per token) + top-1 bucket compaction
// ---------------------------------------------------------------------------
__global__ __launch_bounds__(256) void moe_gate_kernel(
    const float* __restrict__ x, const float* __restrict__ Wg,
    const float* __restrict__ bg, float* __restrict__ gates,
    float* __restrict__ topw, int* __restrict__ counts,
    int* __restrict__ buckets) {
  const int wid  = threadIdx.x >> 5;
  const int lane = threadIdx.x & 31;
  const int tok  = blockIdx.x * 8 + wid;
  if (tok >= kN) return;

  const float* xrow = x + (size_t)tok * kD;
  float acc[kE];
#pragma unroll
  for (int e = 0; e < kE; ++e) acc[e] = 0.0f;

  for (int d = lane; d < kD; d += 32) {
    const float xv = xrow[d];
    const float* wr = Wg + (size_t)d * kE;
#pragma unroll
    for (int e = 0; e < kE; ++e) acc[e] += xv * wr[e];
  }
#pragma unroll
  for (int off = 16; off >= 1; off >>= 1) {
#pragma unroll
    for (int e = 0; e < kE; ++e) acc[e] += __shfl_xor(acc[e], off, 32);
  }

  if (lane == 0) {
    float mx = -1e30f;
#pragma unroll
    for (int e = 0; e < kE; ++e) { acc[e] += bg[e]; mx = fmaxf(mx, acc[e]); }
    float p[kE];
    float s = 0.0f;
#pragma unroll
    for (int e = 0; e < kE; ++e) { p[e] = __expf(acc[e] - mx); s += p[e]; }
    const float inv = 1.0f / s;
    int   bi = 0;
    float bw = -1.0f;
#pragma unroll
    for (int e = 0; e < kE; ++e) {
      const float g = p[e] * inv;
      gates[(size_t)tok * kE + e] = g;
      if (g > bw) { bw = g; bi = e; }
    }
    topw[tok] = bw;
    const int pos = atomicAdd(&counts[bi], 1);
    buckets[(size_t)bi * kN + pos] = tok;
  }
}

// ---------------------------------------------------------------------------
// Kernel A: elementwise fp32 -> bf16 (x)
// ---------------------------------------------------------------------------
__global__ __launch_bounds__(256) void convert_x_bf16(const float* __restrict__ x,
                                                      __bf16* __restrict__ xb) {
  const size_t i = ((size_t)blockIdx.x * blockDim.x + threadIdx.x) * 8;
#pragma unroll
  for (int j = 0; j < 8; ++j) xb[i + j] = (__bf16)x[i + j];
}

// ---------------------------------------------------------------------------
// Kernel B: W [E][R][C] fp32 -> Wt [E][C][R] bf16 (tiled transpose via LDS)
//   block (32,8), grid (C/32, R/32, E)
// ---------------------------------------------------------------------------
__global__ __launch_bounds__(256) void transpose_to_bf16(const float* __restrict__ W,
                                                         __bf16* __restrict__ Wt,
                                                         int R, int C) {
  __shared__ float tile[32][33];
  const int e = blockIdx.z;
  const float* We  = W + (size_t)e * R * C;
  __bf16*     Wte  = Wt + (size_t)e * R * C;
  const int c0 = blockIdx.x * 32, r0 = blockIdx.y * 32;
  const int tx = threadIdx.x, ty = threadIdx.y;
#pragma unroll
  for (int i = 0; i < 4; ++i)
    tile[ty + i * 8][tx] = We[(size_t)(r0 + ty + i * 8) * C + (c0 + tx)];
  __syncthreads();
#pragma unroll
  for (int i = 0; i < 4; ++i)
    Wte[(size_t)(c0 + ty + i * 8) * R + (r0 + tx)] = (__bf16)tile[tx][ty + i * 8];
}

// ---------------------------------------------------------------------------
// Kernel 2 (fast): routed expert FFN on bf16/transposed operands.
//   grid (kN/MT, kE), 256 threads = 8 wave32.
//   Per 16-token tile: async-stage x rows to LDS, GEMM1 -> bf16 hidden in LDS,
//   GEMM2 -> scaled scatter to out. All frags = two b128 loads.
// ---------------------------------------------------------------------------
__global__ __launch_bounds__(256) void moe_expert_bf16_kernel(
    const __bf16* __restrict__ xb, const __bf16* __restrict__ W1t,
    const __bf16* __restrict__ W2t, const float* __restrict__ b1,
    const float* __restrict__ b2, const int* __restrict__ counts,
    const int* __restrict__ buckets, const float* __restrict__ topw,
    float* __restrict__ out) {
  __shared__ __bf16 s_h[16 * HPAD];   // 64.3 KB hidden tile
  __shared__ __bf16 s_x[16 * DPAD];   // 32.3 KB x tile
  __shared__ int    s_tok[MT];
  __shared__ float  s_w[MT];

  const int e    = blockIdx.y;
  const int cnt  = counts[e];
  const int base = blockIdx.x * MT;
  if (base >= cnt) return;  // uniform: EXEC all-ones at every WMMA

  const int tid = threadIdx.x;
  if (tid < MT) {
    const int t   = base + tid;
    const int tok = (t < cnt) ? buckets[(size_t)e * kN + t]
                              : buckets[(size_t)e * kN + base];
    s_tok[tid] = tok;
    s_w[tid]   = (t < cnt) ? topw[tok] : 0.0f;
  }
  __syncthreads();

  const int wv   = tid >> 5;
  const int lane = tid & 31;
  const int nloc = lane & 15;
  const int hi8  = (lane >> 4) * 8;

  const __bf16* W1te = W1t + (size_t)e * kH * kD;  // [H][D]
  const __bf16* W2te = W2t + (size_t)e * kD * kH;  // [D][H]

  for (int mt = 0; mt < MT / 16; ++mt) {
    // ---- stage 16 token rows of xb into LDS (async-to-LDS path) ----
    for (int i = tid; i < 16 * (kD / 8); i += 256) {
      const int m = i >> 7;
      const int c = (i & 127) * 8;
      const __bf16* src = xb + (size_t)s_tok[mt * 16 + m] * kD + c;
      __bf16*       dst = &s_x[m * DPAD + c];
#ifdef ASYNC_LDS
      __builtin_amdgcn_global_load_async_to_lds_b128(
          (AS1 i32x4*)src, (AS3 i32x4*)dst, 0, 0);
#else
      *(uint4*)dst = *(const uint4*)src;
#endif
    }
#ifdef ASYNC_LDS
    asm volatile("s_wait_asynccnt 0x0" ::: "memory");
#endif
    __syncthreads();  // x staged; also: prev tile's GEMM2 done before hidden reuse

    // ---- GEMM1: hidden[16, H] = relu(X @ W1 + b1); wave wv owns 256 H-cols ----
    const __bf16* arow = &s_x[nloc * DPAD];
    for (int ng = 0; ng < 4; ++ng) {
      v8f acc[4] = {};
      for (int kk = 0; kk < kD; kk += 32) {
        const v16bf a = load_frag_bf16(arow + kk + hi8);
#pragma unroll
        for (int q = 0; q < 4; ++q) {
          const int h0 = wv * 256 + (ng * 4 + q) * 16;
          const __bf16* wcol = W1te + (size_t)(h0 + nloc) * kD + kk + hi8;
          __builtin_prefetch(wcol + 64, 0, 1);
          const v16bf b = load_frag_bf16(wcol);
          acc[q] = __builtin_amdgcn_wmma_f32_16x16x32_bf16(
              false, a, false, b, (short)0, acc[q], false, false);
        }
      }
#pragma unroll
      for (int q = 0; q < 4; ++q) {
        const int h0   = wv * 256 + (ng * 4 + q) * 16;
        const float bb = b1[(size_t)e * kH + h0 + nloc];
#pragma unroll
        for (int r = 0; r < 8; ++r) {  // C/D: M = hi*8 + r, N = lane&15
          float v = acc[q][r] + bb;
          v = v > 0.0f ? v : 0.0f;
          s_h[(hi8 + r) * HPAD + h0 + nloc] = (__bf16)v;
        }
      }
    }
    __syncthreads();

    // ---- GEMM2: out[16, D] = gate * (hidden @ W2 + b2); wave wv owns 128 D-cols --
    const __bf16* hrow = &s_h[nloc * HPAD];
    for (int ng = 0; ng < 2; ++ng) {
      v8f acc[4] = {};
      for (int kk = 0; kk < kH; kk += 32) {
        const v16bf a = load_frag_bf16(hrow + kk + hi8);
#pragma unroll
        for (int q = 0; q < 4; ++q) {
          const int d0 = wv * 128 + (ng * 4 + q) * 16;
          const __bf16* wcol = W2te + (size_t)(d0 + nloc) * kH + kk + hi8;
          __builtin_prefetch(wcol + 64, 0, 1);
          const v16bf b = load_frag_bf16(wcol);
          acc[q] = __builtin_amdgcn_wmma_f32_16x16x32_bf16(
              false, a, false, b, (short)0, acc[q], false, false);
        }
      }
#pragma unroll
      for (int q = 0; q < 4; ++q) {
        const int d0   = wv * 128 + (ng * 4 + q) * 16;
        const float bb = b2[(size_t)e * kD + d0 + nloc];
#pragma unroll
        for (int r = 0; r < 8; ++r) {
          const int mm = mt * 16 + hi8 + r;
          if (base + mm < cnt) {
            out[(size_t)s_tok[mm] * kD + d0 + nloc] = (acc[q][r] + bb) * s_w[mm];
          }
        }
      }
    }
    __syncthreads();  // hidden consumed before next tile overwrites
  }
}

// ---------------------------------------------------------------------------
// Kernel 2 (fallback, fp32-from-global) — used only if workspace is too small
// ---------------------------------------------------------------------------
__global__ __launch_bounds__(256) void moe_expert_f32_kernel(
    const float* __restrict__ x, const float* __restrict__ W1,
    const float* __restrict__ b1, const float* __restrict__ W2,
    const float* __restrict__ b2, const int* __restrict__ counts,
    const int* __restrict__ buckets, const float* __restrict__ topw,
    float* __restrict__ out) {
  __shared__ __bf16 s_hidden[MT * HPAD];
  __shared__ int    s_tok[MT];
  __shared__ float  s_w[MT];

  const int e    = blockIdx.y;
  const int cnt  = counts[e];
  const int base = blockIdx.x * MT;
  if (base >= cnt) return;

  const int tid = threadIdx.x;
  if (tid < MT) {
    const int t   = base + tid;
    const int tok = (t < cnt) ? buckets[(size_t)e * kN + t]
                              : buckets[(size_t)e * kN + base];
    s_tok[tid] = tok;
    s_w[tid]   = (t < cnt) ? topw[tok] : 0.0f;
  }
  __syncthreads();

  const int wv   = tid >> 5;
  const int lane = tid & 31;
  const int nloc = lane & 15;
  const int hi8  = (lane >> 4) * 8;

  const float* W1e = W1 + (size_t)e * kD * kH;
  const float* W2e = W2 + (size_t)e * kH * kD;

  for (int mt = 0; mt < MT / 16; ++mt) {
    const float* xrow = x + (size_t)s_tok[mt * 16 + nloc] * kD;
    for (int ng = 0; ng < 4; ++ng) {
      v8f acc[4] = {};
      for (int kk = 0; kk < kD; kk += 32) {
        const v16bf a = load_a_frag_f32(xrow + kk + hi8);
#pragma unroll
        for (int q = 0; q < 4; ++q) {
          const int h0 = wv * 256 + (ng * 4 + q) * 16;
          const v16bf b = load_b_frag_f32(
              W1e + (size_t)(kk + hi8) * kH + (h0 + nloc), kH);
          acc[q] = __builtin_amdgcn_wmma_f32_16x16x32_bf16(
              false, a, false, b, (short)0, acc[q], false, false);
        }
      }
#pragma unroll
      for (int q = 0; q < 4; ++q) {
        const int h0   = wv * 256 + (ng * 4 + q) * 16;
        const float bb = b1[(size_t)e * kH + h0 + nloc];
#pragma unroll
        for (int r = 0; r < 8; ++r) {
          float v = acc[q][r] + bb;
          v = v > 0.0f ? v : 0.0f;
          s_hidden[(mt * 16 + hi8 + r) * HPAD + h0 + nloc] = (__bf16)v;
        }
      }
    }
  }
  __syncthreads();

  for (int mt = 0; mt < MT / 16; ++mt) {
    const __bf16* hrow = &s_hidden[(mt * 16 + nloc) * HPAD];
    for (int ng = 0; ng < 2; ++ng) {
      v8f acc[4] = {};
      for (int kk = 0; kk < kH; kk += 32) {
        const v16bf a = load_frag_bf16(hrow + kk + hi8);
#pragma unroll
        for (int q = 0; q < 4; ++q) {
          const int d0 = wv * 128 + (ng * 4 + q) * 16;
          const v16bf b = load_b_frag_f32(
              W2e + (size_t)(kk + hi8) * kD + (d0 + nloc), kD);
          acc[q] = __builtin_amdgcn_wmma_f32_16x16x32_bf16(
              false, a, false, b, (short)0, acc[q], false, false);
        }
      }
#pragma unroll
      for (int q = 0; q < 4; ++q) {
        const int d0   = wv * 128 + (ng * 4 + q) * 16;
        const float bb = b2[(size_t)e * kD + d0 + nloc];
#pragma unroll
        for (int r = 0; r < 8; ++r) {
          const int mm = mt * 16 + hi8 + r;
          if (base + mm < cnt) {
            out[(size_t)s_tok[mm] * kD + d0 + nloc] = (acc[q][r] + bb) * s_w[mm];
          }
        }
      }
    }
  }
}

// ---------------------------------------------------------------------------
// Host launch. d_out = [out (kN*kD f32) | gate_weights (kN*kE f32)]
// ---------------------------------------------------------------------------
extern "C" void kernel_launch(void* const* d_in, const int* in_sizes, int n_in,
                              void* d_out, int out_size, void* d_ws, size_t ws_size,
                              hipStream_t stream) {
  (void)in_sizes; (void)n_in; (void)out_size;

  const float* x  = (const float*)d_in[0];
  const float* Wg = (const float*)d_in[1];
  const float* bg = (const float*)d_in[2];
  const float* W1 = (const float*)d_in[3];
  const float* b1 = (const float*)d_in[4];
  const float* W2 = (const float*)d_in[5];
  const float* b2 = (const float*)d_in[6];

  float* out   = (float*)d_out;
  float* gates = out + (size_t)kN * kD;

  char*  ws      = (char*)d_ws;
  int*   counts  = (int*)ws;
  float* topw    = (float*)(ws + 256);
  int*   buckets = (int*)(ws + 256 + (size_t)kN * 4);

  moe_init_kernel<<<dim3(1), dim3(32), 0, stream>>>(counts);
  moe_gate_kernel<<<dim3(kN / 8), dim3(256), 0, stream>>>(x, Wg, bg, gates, topw,
                                                          counts, buckets);

  if (ws_size >= WS_TOTAL) {
    __bf16* xb  = (__bf16*)(ws + WS_META);
    __bf16* W1t = (__bf16*)(ws + WS_META + WS_XB);
    __bf16* W2t = (__bf16*)(ws + WS_META + WS_XB + WS_W);

    convert_x_bf16<<<dim3((kN * (size_t)kD) / (256 * 8)), dim3(256), 0, stream>>>(x, xb);
    transpose_to_bf16<<<dim3(kH / 32, kD / 32, kE), dim3(32, 8), 0, stream>>>(
        W1, W1t, kD, kH);  // [D][H] -> [H][D]
    transpose_to_bf16<<<dim3(kD / 32, kH / 32, kE), dim3(32, 8), 0, stream>>>(
        W2, W2t, kH, kD);  // [H][D] -> [D][H]

    moe_expert_bf16_kernel<<<dim3(kN / MT, kE), dim3(256), 0, stream>>>(
        xb, W1t, W2t, b1, b2, counts, buckets, topw, out);
  } else {
    moe_expert_f32_kernel<<<dim3(kN / MT, kE), dim3(256), 0, stream>>>(
        x, W1, b1, W2, b2, counts, buckets, topw, out);
  }
}
